// LinearLIF_29824252903750
// MI455X (gfx1250) — matrix operand order repaired and checked by reading, not compiled, not run on hardware
//
#include <hip/hip_runtime.h>

#define T_STEPS 512
#define BATCH   64
#define DIN     512
#define HID     1024
#define BH      (BATCH*HID)          // 65536
#define TBH     ((size_t)T_STEPS*BH) // 33,554,432

typedef __attribute__((ext_vector_type(16))) __bf16 v16bf;
typedef __attribute__((ext_vector_type(8)))  float  v8f;

union BV {
  v16bf v;
  uint4 q[2];
  unsigned int d[8];
  unsigned short s[16];
};

__device__ __forceinline__ unsigned short f2bf(float f) {
  unsigned int u = __float_as_uint(f);
  unsigned int r = u + 0x7FFFu + ((u >> 16) & 1u);   // round-to-nearest-even
  return (unsigned short)(r >> 16);
}
__device__ __forceinline__ float bf2f(unsigned short h) {
  return __uint_as_float(((unsigned int)h) << 16);
}

__device__ __forceinline__ v8f wmma_bf16(v16bf a, v16bf b, v8f c) {
  // D = A*B + C, f32 accum, 16x16x32 bf16
  return __builtin_amdgcn_wmma_f32_16x16x32_bf16(false, a, false, b, (short)0, c, false, false);
}

// Load 16 f32 (A-operand lane layout: K = kb..kb+7, kb+16..kb+23) and
// split into hi/lo bf16 via truncation + v_perm packing (cheap VALU).
__device__ __forceinline__ void split_pack_A(const float* __restrict__ xr, int kb,
                                             BV& hi, BV& lo) {
  union { float4 q[4]; float f[16]; unsigned int u[16]; } xa;
  xa.q[0] = *(const float4*)(xr + kb);
  xa.q[1] = *(const float4*)(xr + kb + 4);
  xa.q[2] = *(const float4*)(xr + kb + 16);
  xa.q[3] = *(const float4*)(xr + kb + 20);
#pragma unroll
  for (int p = 0; p < 8; ++p) {
    unsigned int u0 = xa.u[2 * p], u1 = xa.u[2 * p + 1];
    hi.d[p] = __builtin_amdgcn_perm(u1, u0, 0x07060302u);      // top halves -> bf16 pair
    float l0 = xa.f[2 * p]     - __uint_as_float(u0 & 0xFFFF0000u);
    float l1 = xa.f[2 * p + 1] - __uint_as_float(u1 & 0xFFFF0000u);
    lo.d[p] = __builtin_amdgcn_perm(__float_as_uint(l1), __float_as_uint(l0), 0x07060302u);
  }
}

// ---- workspace layout (ushort elems) ----
// W_in packed:  [ntg 64][mat 2][chunk 16][lane 32][e 16]  = 1,048,576
// W_rec packed: [wg 16][nt 4][mat 2][chunk 32][lane 32][e 16] = 2,097,152
// spike double buffer: 2 * B*H bf16 = 131,072 ; then a u32 barrier counter
#define WIN_PK   1048576
#define WREC_PK  2097152
#define SBUF_N   131072

// =====================================================================
// Prep: split f32 weights into hi/lo bf16 in WMMA B-operand lane layout,
// zero spike buffers + barrier counter.  B mapping within a 16-col tile:
//   N = (lane&15), K = chunk*32 + (lane>>4)*8 + (e&7) + (e>>3)*16
// =====================================================================
__global__ __launch_bounds__(256) void lif_prep(const float* __restrict__ Win,
                                                const float* __restrict__ Wrec,
                                                unsigned short* __restrict__ wpk_in,
                                                unsigned short* __restrict__ wpk_rec,
                                                unsigned short* __restrict__ sbuf,
                                                unsigned int* __restrict__ ctr) {
  int i = blockIdx.x * 256 + threadIdx.x;
  if (i < WIN_PK) {
    int e     = i & 15;
    int lane  = (i >> 4) & 31;
    int chunk = (i >> 9) & 15;
    int mat   = (i >> 13) & 1;
    int ntg   = i >> 14;                 // 0..63
    int n = ntg * 16 + (lane & 15);
    int k = chunk * 32 + (lane >> 4) * 8 + (e & 7) + (e >> 3) * 16;
    float w = Win[n * DIN + k];
    unsigned short hi = f2bf(w);
    wpk_in[i] = mat ? f2bf(w - bf2f(hi)) : hi;
  } else if (i < WIN_PK + WREC_PK) {
    int j = i - WIN_PK;
    int e     = j & 15;
    int lane  = (j >> 4) & 31;
    int chunk = (j >> 9) & 31;
    int mat   = (j >> 14) & 1;
    int nt    = (j >> 15) & 3;
    int wg    = j >> 17;                 // 0..15
    int n = wg * 64 + nt * 16 + (lane & 15);
    int k = chunk * 32 + (lane >> 4) * 8 + (e & 7) + (e >> 3) * 16;
    float w = Wrec[n * HID + k];
    unsigned short hi = f2bf(w);
    wpk_rec[j] = mat ? f2bf(w - bf2f(hi)) : hi;
  } else if (i < WIN_PK + WREC_PK + SBUF_N) {
    sbuf[i - WIN_PK - WREC_PK] = 0;      // s[-1] = 0
  } else if (i == WIN_PK + WREC_PK + SBUF_N) {
    *ctr = 0u;
  }
}

// =====================================================================
// Phase 1: xproj[T*B, H] = X[T*B, DIN] @ W_in.T  (split-bf16).
// Block = 8 waves covering M=64 x N=256. Each wave: 2 M-tiles x 4 N-tiles
// (8 accumulators), so A conversion is amortized over 12 WMMAs and the
// two M-tile accumulators interleave to hide WMMA->WMMA latency.
// grid = (32768/64, 1024/256) = (512, 4)
// =====================================================================
__global__ __launch_bounds__(256) void lif_xproj(const float* __restrict__ x,
                                                 const unsigned short* __restrict__ wpk_in,
                                                 float* __restrict__ hid) {
  int lane = threadIdx.x & 31;
  int w    = threadIdx.x >> 5;
  int m0   = blockIdx.x * 64 + (w >> 2) * 32;     // this wave: M-tiles m0, m0+16
  int ntg0 = blockIdx.y * 16 + (w & 3) * 4;       // this wave: 4 N-tiles

  const float* xr0 = x + (size_t)(m0 + (lane & 15)) * DIN;
  const float* xr1 = xr0 + (size_t)16 * DIN;

  v8f acc[2][4];
#pragma unroll
  for (int a = 0; a < 2; ++a)
#pragma unroll
    for (int nt = 0; nt < 4; ++nt) acc[a][nt] = (v8f){};

  for (int chunk = 0; chunk < 16; ++chunk) {
    int kb = chunk * 32 + (lane >> 4) * 8;
    BV ahi0, alo0, ahi1, alo1;
    split_pack_A(xr0, kb, ahi0, alo0);
    split_pack_A(xr1, kb, ahi1, alo1);

#pragma unroll
    for (int nt = 0; nt < 4; ++nt) {
      int eb = (ntg0 + nt) * 16384 + chunk * 512 + (lane << 4);
      const uint4* bp  = (const uint4*)(wpk_in + eb);          // hi
      const uint4* bpl = (const uint4*)(wpk_in + eb + 8192);   // lo
      BV bhi, blo;
      bhi.q[0] = bp[0];  bhi.q[1] = bp[1];
      blo.q[0] = bpl[0]; blo.q[1] = bpl[1];
      // interleave the two M-tile accumulators
      acc[0][nt] = wmma_bf16(ahi0.v, bhi.v, acc[0][nt]);
      acc[1][nt] = wmma_bf16(ahi1.v, bhi.v, acc[1][nt]);
      acc[0][nt] = wmma_bf16(alo0.v, bhi.v, acc[0][nt]);
      acc[1][nt] = wmma_bf16(alo1.v, bhi.v, acc[1][nt]);
      acc[0][nt] = wmma_bf16(ahi0.v, blo.v, acc[0][nt]);
      acc[1][nt] = wmma_bf16(ahi1.v, blo.v, acc[1][nt]);
    }
  }
  // C layout: element v of lane L -> row +v + (L/16)*8, col + L%16
#pragma unroll
  for (int a = 0; a < 2; ++a) {
    int rbase = m0 + a * 16 + (lane >> 4) * 8;
#pragma unroll
    for (int nt = 0; nt < 4; ++nt) {
      int col = (ntg0 + nt) * 16 + (lane & 15);
#pragma unroll
      for (int v = 0; v < 8; ++v)
        hid[(size_t)(rbase + v) * HID + col] = acc[a][nt][v];
    }
  }
}

// =====================================================================
// Phase 2: persistent sequential scan. 16 WGs x 256 threads (8 waves).
// WG wg owns H columns [wg*64, wg*64+64). W_rec slice (hi+lo bf16,
// 256 KB) resident in LDS for all 512 steps. Wave w: M-tile w>>1,
// two N-tiles (w&1)*2,(w&1)*2+1. h[t-1] kept in VGPR accumulators.
// =====================================================================
__global__ __launch_bounds__(256) void lif_scan(const unsigned short* __restrict__ wpk_rec,
                                                unsigned short* __restrict__ sbuf,
                                                float* __restrict__ spk,
                                                float* __restrict__ hid,
                                                const float* __restrict__ leak_p,
                                                const float* __restrict__ vth_p,
                                                unsigned int* __restrict__ ctr) {
  extern __shared__ unsigned short lds_w[];      // 262144 B = WG's packed W_rec slice
  const int wg   = blockIdx.x;
  const int tid  = threadIdx.x;
  const int lane = tid & 31;
  const int w    = tid >> 5;

  const float leak = *leak_p;
  const float vth  = *vth_p;
  const float lvth = leak * vth;

  // Stage this WG's W_rec slice (hi+lo) into LDS once.
  {
    const uint4* src = (const uint4*)(wpk_rec + (size_t)wg * 131072);
    uint4* dst = (uint4*)lds_w;
    for (int i = tid; i < 16384; i += 256) dst[i] = src[i];
  }
  __syncthreads();

  const int m0   = (w >> 1) * 16;                       // batch-tile
  const int np   = (w & 1) * 2;                         // local N-tile pair
  const int colA = wg * 64 + np * 16 + (lane & 15);
  const int colB = colA + 16;
  const int arow = m0 + (lane & 15);
  const int brow = m0 + (lane >> 4) * 8;

  v8f h0 = {}, h1 = {};                                 // resident hidden state

  for (int t = 0; t < T_STEPS; ++t) {
    const unsigned short* S  = sbuf + (t & 1) * BH;
    unsigned short*       Sn = sbuf + ((t + 1) & 1) * BH;

    size_t tb = (size_t)t * BH + (size_t)brow * HID;
#pragma unroll
    for (int v = 0; v < 8; ++v) {                       // pre-pull xproj from HBM/L2
      __builtin_prefetch(&hid[tb + (size_t)v * HID + colA], 0, 0);
      __builtin_prefetch(&hid[tb + (size_t)v * HID + colB], 0, 0);
    }

    v8f a0 = {}, a1 = {};
#pragma unroll 2
    for (int chunk = 0; chunk < 32; ++chunk) {
      // A tile: spikes[t-1], rows m0..m0+15 -- lane L: K = kb..kb+7, kb+16..kb+23
      BV a;
      const uint4* ap = (const uint4*)(S + arow * HID + chunk * 32 + ((lane >> 4) << 3));
      a.q[0] = ap[0];
      a.q[1] = ap[2];                                   // +16 elems = +32 B

      int o00 = np * 32768 + chunk * 512 + (lane << 4);
      BV b00, b01, b10, b11;
      b00.q[0] = *(const uint4*)(lds_w + o00);          // tile0 hi
      b00.q[1] = *(const uint4*)(lds_w + o00 + 8);
      b01.q[0] = *(const uint4*)(lds_w + o00 + 16384);  // tile0 lo
      b01.q[1] = *(const uint4*)(lds_w + o00 + 16392);
      b10.q[0] = *(const uint4*)(lds_w + o00 + 32768);  // tile1 hi
      b10.q[1] = *(const uint4*)(lds_w + o00 + 32776);
      b11.q[0] = *(const uint4*)(lds_w + o00 + 49152);  // tile1 lo
      b11.q[1] = *(const uint4*)(lds_w + o00 + 49160);
      // interleave accumulator chains (a0,a1,a0,a1)
      a0 = wmma_bf16(a.v, b00.v, a0);
      a1 = wmma_bf16(a.v, b10.v, a1);
      a0 = wmma_bf16(a.v, b01.v, a0);
      a1 = wmma_bf16(a.v, b11.v, a1);
    }

    // elementwise LIF update + emit outputs + next-step bf16 spikes
#pragma unroll
    for (int v = 0; v < 8; ++v) {
      size_t ro = tb + (size_t)v * HID;
      {
        float xp = hid[ro + colA];
        float hp = h0[v];
        float sp = (hp >= vth) ? 1.0f : 0.0f;
        float h  = xp + a0[v] + leak * hp - sp * lvth;
        float sv = (h >= vth) ? 1.0f : 0.0f;
        hid[ro + colA] = h;
        spk[ro + colA] = sv;
        Sn[(brow + v) * HID + colA] = (h >= vth) ? (unsigned short)0x3F80 : (unsigned short)0;
        h0[v] = h;
      }
      {
        float xp = hid[ro + colB];
        float hp = h1[v];
        float sp = (hp >= vth) ? 1.0f : 0.0f;
        float h  = xp + a1[v] + leak * hp - sp * lvth;
        float sv = (h >= vth) ? 1.0f : 0.0f;
        hid[ro + colB] = h;
        spk[ro + colB] = sv;
        Sn[(brow + v) * HID + colB] = (h >= vth) ? (unsigned short)0x3F80 : (unsigned short)0;
        h1[v] = h;
      }
    }

    // ---- inter-workgroup step barrier ----
    __threadfence();                                    // release spike stores (agent scope)
    __syncthreads();
    if (tid == 0)
      __hip_atomic_fetch_add(ctr, 1u, __ATOMIC_RELEASE, __HIP_MEMORY_SCOPE_AGENT);
    __builtin_amdgcn_s_cluster_barrier();               // fast path if cluster-dispatched; NOP otherwise
    if (tid == 0) {
      unsigned target = (unsigned)(t + 1) * 16u;
      while (__hip_atomic_load(ctr, __ATOMIC_ACQUIRE, __HIP_MEMORY_SCOPE_AGENT) < target)
        __builtin_amdgcn_s_sleep(1);
    }
    __syncthreads();
    __builtin_amdgcn_fence(__ATOMIC_ACQUIRE, "agent");  // invalidate L0 before reading new spikes
  }
}

// =====================================================================
extern "C" void kernel_launch(void* const* d_in, const int* in_sizes, int n_in,
                              void* d_out, int out_size, void* d_ws, size_t ws_size,
                              hipStream_t stream) {
  const float* x     = (const float*)d_in[0];
  const float* W_in  = (const float*)d_in[1];
  const float* W_rec = (const float*)d_in[2];
  const float* leak  = (const float*)d_in[3];
  const float* vth   = (const float*)d_in[4];

  float* spk = (float*)d_out;
  float* hid = spk + TBH;

  unsigned short* ws16    = (unsigned short*)d_ws;
  unsigned short* wpk_in  = ws16;
  unsigned short* wpk_rec = ws16 + WIN_PK;
  unsigned short* sbuf    = ws16 + WIN_PK + WREC_PK;
  unsigned int*   ctr     = (unsigned int*)((char*)d_ws +
                            (size_t)2 * (WIN_PK + WREC_PK + SBUF_N));

  const int prep_total = WIN_PK + WREC_PK + SBUF_N + 1;
  lif_prep<<<(prep_total + 255) / 256, 256, 0, stream>>>(W_in, W_rec, wpk_in, wpk_rec, sbuf, ctr);

  lif_xproj<<<dim3(512, 4), 256, 0, stream>>>(x, wpk_in, hid);

  lif_scan<<<16, 256, 262144, stream>>>(wpk_rec, sbuf, spk, hid, leak, vth, ctr);
}
